// ObjectAwareESPCN_32822140076770
// MI455X (gfx1250) — matrix operand order repaired and checked by reading, not compiled, not run on hardware
//
#include <hip/hip_runtime.h>
#include <math.h>

// ---------------------------------------------------------------------------
// Fused MoE-ESPCN for MI455X (gfx1250, wave32, WMMA).
// Per image, run ONLY the selected expert (one-hot masking == expert select).
// One workgroup = one 16x16 pre-shuffle tile of one image; h1/h2 live in LDS.
// conv1: VALU (K=25 too small for WMMA). conv2/conv3: implicit GEMM via
// v_wmma_f32_16x16x32_f16 with multiple M-tiles in flight per wave so the
// XDL pipe is not serialized on LDS latency, and B fragments are loaded
// once per wave and reused across M-tiles.
// ---------------------------------------------------------------------------

typedef __attribute__((ext_vector_type(16))) _Float16 v16h;
typedef __attribute__((ext_vector_type(8)))  _Float16 v8h;
typedef __attribute__((ext_vector_type(8)))  float    v8f;

// LDS layout (bytes):
//   s_in  : 24*24 f32          =  2304   @ 0
//   s_w1  : 64*25 f32          =  6400   @ 2304
//   s_b1  : 64 f32             =   256   @ 8704
//   s_h1  : 20*20*64 f16       = 51200   @ 8960
//   s_h2  : 18*18*32 f16       = 20736   @ 60160
//   s_bw2 : 36 frags * 512 f16 = 36864   @ 80896   (tap*4 + kc*2 + ntile)
//   s_bw3 : 9  frags * 512 f16 =  9216   @ 117760
// total = 126976 B  (dynamic LDS; 2 WGs per 320KB WGP)
#define SMEM_BYTES 126976

#define SHUF16(lo, hi) __builtin_shufflevector(lo, hi, 0,1,2,3,4,5,6,7, \
                                               8,9,10,11,12,13,14,15)

__global__ __launch_bounds__(256)
void espcn_moe_fused(const float* __restrict__ xg,
                     const int*   __restrict__ cig,
                     const float* __restrict__ w1g, const float* __restrict__ b1g,
                     const float* __restrict__ w2g, const float* __restrict__ b2g,
                     const float* __restrict__ w3g, const float* __restrict__ b3g,
                     float* __restrict__ outg)
{
    extern __shared__ char smem[];
    float*    s_in  = (float*)(smem);
    float*    s_w1  = (float*)(smem + 2304);
    float*    s_b1  = (float*)(smem + 8704);
    _Float16* s_h1  = (_Float16*)(smem + 8960);
    _Float16* s_h2  = (_Float16*)(smem + 60160);
    _Float16* s_bw2 = (_Float16*)(smem + 80896);
    _Float16* s_bw3 = (_Float16*)(smem + 117760);

    const int tid  = threadIdx.x;
    const int img  = blockIdx.y;
    const int tile = blockIdx.x;
    const int ty   = (tile >> 3) * 16;   // tile row origin (pre-shuffle coords)
    const int tx   = (tile & 7)  * 16;

    const int e = cig[img];              // expert for this image

    const float* xim = xg  + (size_t)img * 128 * 128;
    const float* w1e = w1g + (size_t)e * 64 * 25;
    const float* b1e = b1g + (size_t)e * 64;
    const float* w2e = w2g + (size_t)e * 32 * 64 * 9;
    const float* b2e = b2g + (size_t)e * 32;
    const float* w3e = w3g + (size_t)e * 9 * 32 * 9;
    const float* b3e = b3g + (size_t)e * 9;

    // ---- Stage input tile 24x24 (origin ty-4, tx-4), zero padded ----------
    for (int idx = tid; idx < 576; idx += 256) {
        int iy = idx / 24, ix = idx - iy * 24;
        int gy = ty - 4 + iy, gx = tx - 4 + ix;
        float v = 0.0f;
        if (gy >= 0 && gy < 128 && gx >= 0 && gx < 128) v = xim[gy * 128 + gx];
        s_in[idx] = v;
    }
    // ---- Stage conv1 weights/bias (f32) -----------------------------------
    for (int idx = tid; idx < 1600; idx += 256) s_w1[idx] = w1e[idx];
    if (tid < 64) s_b1[tid] = b1e[tid];

    // ---- Stage conv2 weights as WMMA B fragments (f16) --------------------
    // B 32x16 f16 layout: lane holds column n = lane%16; halves i hold
    // k = (i&7) + (i>=8 ? 16 : 0) + (lane>=16 ? 8 : 0).  frag = tap*4 + kc*2 + ntile.
    for (int idx = tid; idx < 36 * 512; idx += 256) {
        int frag = idx >> 9;
        int r    = idx & 511;
        int ln   = r >> 4;
        int i    = r & 15;
        int tap  = frag >> 2;
        int kc   = (frag >> 1) & 1;
        int nt   = frag & 1;
        int co   = nt * 16 + (ln & 15);
        int ci   = kc * 32 + (i & 7) + ((i >> 3) << 4) + ((ln >= 16) ? 8 : 0);
        s_bw2[idx] = (_Float16)w2e[(co * 64 + ci) * 9 + tap];
    }
    // ---- Stage conv3 weights as WMMA B fragments (f16), N padded 9->16 ----
    for (int idx = tid; idx < 9 * 512; idx += 256) {
        int tap = idx >> 9;
        int r   = idx & 511;
        int ln  = r >> 4;
        int i   = r & 15;
        int co  = ln & 15;
        int ci  = (i & 7) + ((i >> 3) << 4) + ((ln >= 16) ? 8 : 0);
        _Float16 v = (_Float16)0.0f;
        if (co < 9) v = (_Float16)w3e[(co * 32 + ci) * 9 + tap];
        s_bw3[idx] = v;
    }
    __syncthreads();

    // ---- Phase 1: conv1 5x5 (1->64) + tanh, VALU, h1 is 20x20x64 f16 ------
    for (int p = tid; p < 400; p += 256) {
        int py = p / 20, px = p - py * 20;
        float pin[25];
        #pragma unroll
        for (int r = 0; r < 5; ++r)
            #pragma unroll
            for (int c = 0; c < 5; ++c)
                pin[r * 5 + c] = s_in[(py + r) * 24 + (px + c)];
        for (int ch = 0; ch < 64; ++ch) {
            float acc = s_b1[ch];
            const float* wv = &s_w1[ch * 25];
            #pragma unroll
            for (int j = 0; j < 25; ++j) acc = fmaf(pin[j], wv[j], acc);
            s_h1[p * 64 + ch] = (_Float16)tanhf(acc);
        }
    }
    __syncthreads();

    const int wave = tid >> 5;
    const int lane = tid & 31;
    const int lhal = (lane >= 16) ? 1 : 0;   // which half-lane group
    const int lmod = lane & 15;

    // ---- Phase 2: conv2 3x3 (64->32) + tanh via WMMA ----------------------
    // Implicit GEMM: M = 324 pixels (18x18), N = 32, K = 9 taps x 64 ch.
    // Each wave keeps its 3 M-tiles (wave, wave+8, wave+16) in flight:
    // 6 independent accumulators, each B fragment loaded once per wave.
    {
        int mtv[3];
        mtv[0] = wave;
        mtv[1] = wave + 8;
        const bool t2v = (wave + 16) < 21;
        mtv[2] = t2v ? (wave + 16) : wave;   // duplicate work, writes skipped
                                             // in the epilogue when !t2v
        int base[3];
        #pragma unroll
        for (int t = 0; t < 3; ++t) {
            int m  = mtv[t] * 16 + lmod;
            int mc = (m < 324) ? m : 0;      // pad rows read pixel 0
            int py2 = mc / 18, px2 = mc - py2 * 18;
            base[t] = py2 * 20 + px2;        // h1 pixel index (20-wide grid)
        }
        v8f acc[3][2];
        #pragma unroll
        for (int t = 0; t < 3; ++t) { acc[t][0] = (v8f){}; acc[t][1] = (v8f){}; }

        #pragma unroll
        for (int tap = 0; tap < 9; ++tap) {
            int dy = tap / 3, dx = tap - dy * 3;
            int poff = dy * 20 + dx;
            #pragma unroll
            for (int kc = 0; kc < 2; ++kc) {
                // B fragments: shared by all 3 M-tiles of this wave.
                const _Float16* bp = s_bw2 + ((tap * 4 + kc * 2) * 32 + lane) * 16;
                v16h b0 = *(const v16h*)(bp);
                v16h b1 = *(const v16h*)(bp + 512);
                // A 16x32 f16 layout: halves 0..7 -> K=cb..cb+7,
                // halves 8..15 -> K=cb+16..cb+23, cb = kc*32 + (lane>=16)*8.
                int cb = kc * 32 + lhal * 8;
                #pragma unroll
                for (int t = 0; t < 3; ++t) {
                    const _Float16* hp = s_h1 + (base[t] + poff) * 64;
                    v8h lo = *(const v8h*)(hp + cb);
                    v8h hi = *(const v8h*)(hp + cb + 16);
                    v16h a = SHUF16(lo, hi);
                    acc[t][0] = __builtin_amdgcn_wmma_f32_16x16x32_f16(
                                    false, a, false, b0, (short)0, acc[t][0],
                                    false, false);
                    acc[t][1] = __builtin_amdgcn_wmma_f32_16x16x32_f16(
                                    false, a, false, b1, (short)0, acc[t][1],
                                    false, false);
                }
            }
        }
        // Epilogue: bias + tanh -> h2 (f16). D layout: lane n = lane%16,
        // vgpr j -> M = j + (lane>=16)*8.
        float bias0 = b2e[lmod];
        float bias1 = b2e[lmod + 16];
        #pragma unroll
        for (int t = 0; t < 3; ++t) {
            if (t == 2 && !t2v) break;
            #pragma unroll
            for (int j = 0; j < 8; ++j) {
                int mm = mtv[t] * 16 + j + lhal * 8;
                if (mm < 324) {
                    s_h2[mm * 32 + lmod]      = (_Float16)tanhf(acc[t][0][j] + bias0);
                    s_h2[mm * 32 + lmod + 16] = (_Float16)tanhf(acc[t][1][j] + bias1);
                }
            }
        }
    }
    __syncthreads();

    // ---- Phase 3: conv3 3x3 (32->9) via WMMA + pixel shuffle x3 -----------
    // M-tile = one row (16 px) of the 16x16 tile; N = 16 (9 valid), K = 9x32.
    // Each wave runs rows (wave, wave+8) together, sharing B fragments.
    {
        float* outim = outg + (size_t)img * 384 * 384;
        const int mt0 = wave, mt1 = wave + 8;
        v8f acc0 = {};
        v8f acc1 = {};
        const int cb = lhal * 8;
        #pragma unroll
        for (int tap = 0; tap < 9; ++tap) {
            int dy = tap / 3, dx = tap - dy * 3;
            v16h b = *(const v16h*)(s_bw3 + (tap * 32 + lane) * 16);
            const _Float16* hp0 = s_h2 + ((mt0 + dy) * 18 + (lmod + dx)) * 32;
            const _Float16* hp1 = s_h2 + ((mt1 + dy) * 18 + (lmod + dx)) * 32;
            v8h lo0 = *(const v8h*)(hp0 + cb);
            v8h hi0 = *(const v8h*)(hp0 + cb + 16);
            v8h lo1 = *(const v8h*)(hp1 + cb);
            v8h hi1 = *(const v8h*)(hp1 + cb + 16);
            v16h a0 = SHUF16(lo0, hi0);
            v16h a1 = SHUF16(lo1, hi1);
            acc0 = __builtin_amdgcn_wmma_f32_16x16x32_f16(
                       false, a0, false, b, (short)0, acc0, false, false);
            acc1 = __builtin_amdgcn_wmma_f32_16x16x32_f16(
                       false, a1, false, b, (short)0, acc1, false, false);
        }
        int co = lmod;                        // conv3 output channel
        if (co < 9) {
            float bias = b3e[co];
            int oy = co / 3, ox = co - oy * 3; // pixel-shuffle sub-position
            #pragma unroll
            for (int j = 0; j < 8; ++j) {
                int px = j + lhal * 8;        // D-row -> pixel column
                int gy0 = ty + mt0, gy1 = ty + mt1, gx = tx + px;
                outim[(gy0 * 3 + oy) * 384 + (gx * 3 + ox)] = acc0[j] + bias;
                outim[(gy1 * 3 + oy) * 384 + (gx * 3 + ox)] = acc1[j] + bias;
            }
        }
    }
}

extern "C" void kernel_launch(void* const* d_in, const int* in_sizes, int n_in,
                              void* d_out, int out_size, void* d_ws, size_t ws_size,
                              hipStream_t stream) {
    const float* x  = (const float*)d_in[0];
    const int*   ci = (const int*)  d_in[1];
    const float* w1 = (const float*)d_in[2];
    const float* b1 = (const float*)d_in[3];
    const float* w2 = (const float*)d_in[4];
    const float* b2 = (const float*)d_in[5];
    const float* w3 = (const float*)d_in[6];
    const float* b3 = (const float*)d_in[7];
    float* out = (float*)d_out;

    dim3 grid(64, 32);   // 8x8 tiles of 16x16 per image, 32 images
    espcn_moe_fused<<<grid, 256, SMEM_BYTES, stream>>>(
        x, ci, w1, b1, w2, b2, w3, b3, out);
}
